// LakeDetectionLoss_55216099557589
// MI455X (gfx1250) — compile-verified, matrix operand order checked by reading, not run on hardware
//
#include <hip/hip_runtime.h>
#include <stdint.h>

// LakeDetectionLoss for MI455X (gfx1250, wave32).
// Pipeline per image: init labels -> 1-pass atomic union-find merge ->
// compress (+root flags) -> area + pair-hash stats -> threshold match ->
// WMMA IU8 byte-count reductions -> scalar loss.

#define H_DIM 1024
#define W_DIM 1024
#define HW_N  (H_DIM * W_DIM)      // 1<<20
#define NIMG  8
#define INV_LBL 0xFFFFFFFFu
#define HASH_BITS 21
#define HASH_SIZE (1u << HASH_BITS)
#define HASH_EMPTY 0xFFFFFFFFFFFFFFFFull

typedef int v8i __attribute__((ext_vector_type(8)));

// ---------------- clearing ----------------
__global__ void k_clear_acc(unsigned int* acc) {
  int i = blockIdx.x * blockDim.x + threadIdx.x;
  if (i < 3 * NIMG) acc[i] = 0u;
}

__global__ void k_clear_img(unsigned int* area, unsigned char* matched,
                            unsigned long long* hash) {
  unsigned i = blockIdx.x * blockDim.x + threadIdx.x;
  if (i < HW_N) { area[i] = 0u; matched[i] = 0; }
  if (i < HASH_SIZE) hash[i] = HASH_EMPTY;
}

// ---------------- label init ----------------
__global__ void k_init_labels(const float* __restrict__ logit,
                              const float* __restrict__ target,
                              unsigned int* __restrict__ Lp,
                              unsigned int* __restrict__ Lt) {
  unsigned i = blockIdx.x * blockDim.x + threadIdx.x;
  if (i >= HW_N) return;
  // sigmoid(x) > 0.5  <=>  x > 0
  Lp[i] = (logit[i] > 0.0f) ? i : INV_LBL;
  Lt[i] = (target[i] != 0.0f) ? i : INV_LBL;
}

// ---------------- union-find ----------------
__device__ __forceinline__ unsigned find_root(const unsigned* L, unsigned i) {
  const volatile unsigned* vL = L;
  unsigned v = vL[i];
  while (v != i) { i = v; v = vL[i]; }
  return i;
}

__device__ __forceinline__ void unite(unsigned* L, unsigned a, unsigned b) {
  bool done = false;
  while (!done) {
    a = find_root(L, a);
    b = find_root(L, b);
    if (a < b) {
      unsigned old = atomicMin(&L[b], a);
      done = (old == b);
      b = old;
    } else if (b < a) {
      unsigned old = atomicMin(&L[a], b);
      done = (old == a);
      a = old;
    } else {
      done = true;
    }
  }
}

__global__ void k_merge(unsigned int* L) {
  unsigned i = blockIdx.x * blockDim.x + threadIdx.x;
  if (i >= HW_N) return;
  if (L[i] == INV_LBL) return;
  unsigned x = i & (W_DIM - 1);
  unsigned y = i >> 10;
  if (x + 1 < W_DIM && L[i + 1] != INV_LBL) unite(L, i, i + 1);
  if (y + 1 < H_DIM && L[i + W_DIM] != INV_LBL) unite(L, i, i + W_DIM);
}

__global__ void k_compress(unsigned int* L, unsigned char* rootflag) {
  unsigned i = blockIdx.x * blockDim.x + threadIdx.x;
  if (i >= HW_N) return;
  unsigned v = L[i];
  if (v == INV_LBL) { rootflag[i] = 0; return; }
  unsigned r = find_root(L, i);
  L[i] = r;
  rootflag[i] = (r == i) ? (unsigned char)1 : (unsigned char)0;
}

// ---------------- per-pixel stats ----------------
__global__ void k_stats(const unsigned int* __restrict__ Lt,
                        const unsigned int* __restrict__ Lp,
                        unsigned int* __restrict__ area,
                        unsigned long long* __restrict__ hash) {
  unsigned i = blockIdx.x * blockDim.x + threadIdx.x;
  if (i >= HW_N) return;
  unsigned t = Lt[i];
  if (t == INV_LBL) return;
  atomicAdd(&area[t], 1u);
  unsigned p = Lp[i];
  if (p == INV_LBL) return;
  // key: 40 bits (t:20 | p:20). Entry: key<<24 | count (count <= 2^20 < 2^24).
  unsigned long long key = ((unsigned long long)t << 20) | (unsigned long long)p;
  unsigned long long h = key * 0x9E3779B97F4A7C15ull;
  unsigned slot = (unsigned)(h >> 40) & (HASH_SIZE - 1);
  unsigned long long ins = (key << 24) | 1ull;
  for (;;) {
    unsigned long long cur = hash[slot];
    if (cur == HASH_EMPTY) {
      unsigned long long prev = atomicCAS(&hash[slot], HASH_EMPTY, ins);
      if (prev == HASH_EMPTY) return;  // inserted with count=1
      cur = prev;
    }
    if ((cur >> 24) == key) { atomicAdd(&hash[slot], 1ull); return; }
    slot = (slot + 1) & (HASH_SIZE - 1);
  }
}

// ---------------- matching ----------------
__global__ void k_match(const unsigned long long* __restrict__ hash,
                        const unsigned int* __restrict__ area,
                        unsigned char* __restrict__ matched) {
  unsigned i = blockIdx.x * blockDim.x + threadIdx.x;
  if (i >= HASH_SIZE) return;
  unsigned long long cur = hash[i];
  if (cur == HASH_EMPTY) return;
  unsigned long long key = cur >> 24;
  unsigned cnt = (unsigned)(cur & 0xFFFFFFull);
  unsigned t = (unsigned)(key >> 20);
  float a = (float)area[t];
  float c = (float)cnt;
  if (c > 0.5f * a && c < 1.6f * a) matched[t] = 1;  // benign race, same value
}

// ---------------- WMMA IU8 byte-count reduction ----------------
// Sums HW_N flag bytes using V_WMMA_I32_16X16X64_IU8 with all-ones A.
// Each wmma folds 1024 bytes (B = 64x16 i8 tile, one v8i per lane) into the
// i32 accumulator tile. Sum of the full 16x16 D tile == 16 * (byte total),
// independent of VGPR layout, so acc accumulates 16x the count (final kernel
// shifts right by 4). EXEC stays all-ones: 1024 full waves, one chunk each,
// uniform control flow, HW_N % 1024 == 0 so no tail.
__global__ void k_count_wmma(const unsigned char* __restrict__ bytes,
                             unsigned int* __restrict__ acc) {
  const int lane  = threadIdx.x & 31;
  const int wave  = (int)((blockIdx.x * blockDim.x + threadIdx.x) >> 5);
  const int nwave = (int)((gridDim.x * blockDim.x) >> 5);

  v8i ones;
#pragma unroll
  for (int k = 0; k < 8; ++k) ones[k] = 0x01010101;
  v8i c = {};

  for (long long base = (long long)wave * 1024; base < (long long)HW_N;
       base += (long long)nwave * 1024) {
    const v8i* bp = (const v8i*)(bytes + base);  // 32B-aligned per lane
    v8i b = bp[lane];
    c = __builtin_amdgcn_wmma_i32_16x16x64_iu8(
        /*sgn_a=*/false, ones, /*sgn_b=*/false, b, c,
        /*reuse_a=*/false, /*reuse_b=*/false);
  }
  unsigned s = (unsigned)(c[0] + c[1] + c[2] + c[3] + c[4] + c[5] + c[6] + c[7]);
  atomicAdd(acc, s);  // accumulates 16x the true count
}

// ---------------- final loss ----------------
__global__ void k_final(const unsigned int* __restrict__ acc, float* out) {
  if (blockIdx.x != 0 || threadIdx.x != 0) return;
  float TP = 0.0f, FP = 0.0f, FN = 0.0f;
  for (int i = 0; i < NIMG; ++i) {
    int m = (int)(acc[i] >> 4);             // matched targets
    int t = (int)(acc[NIMG + i] >> 4);      // target components
    int p = (int)(acc[2 * NIMG + i] >> 4);  // pred components
    TP += (float)m;
    int fp = p - m; if (fp < 0) fp = 0;
    int fn = t - m; if (fn < 0) fn = 0;
    FP += (float)fp;
    FN += (float)fn;
  }
  out[0] = 1.0f - (TP + 1.0f) / (TP + FP + FN + 1.0f);
}

// ---------------- launch ----------------
extern "C" void kernel_launch(void* const* d_in, const int* in_sizes, int n_in,
                              void* d_out, int out_size, void* d_ws, size_t ws_size,
                              hipStream_t stream) {
  (void)in_sizes; (void)n_in; (void)out_size; (void)ws_size;
  const float* inputs  = (const float*)d_in[0];  // [8,1,1024,1024] f32 logits
  const float* targets = (const float*)d_in[1];  // [8,1,1024,1024] f32 {0,1}
  float* out = (float*)d_out;

  char* ws = (char*)d_ws;
  size_t off = 0;
  auto alloc = [&](size_t bytes) -> void* {
    void* p = (void*)(ws + off);
    off += (bytes + 255) & ~(size_t)255;
    return p;
  };
  unsigned int*       acc     = (unsigned int*)alloc(3 * NIMG * sizeof(unsigned));
  unsigned int*       Lt      = (unsigned int*)alloc((size_t)HW_N * 4);
  unsigned int*       Lp      = (unsigned int*)alloc((size_t)HW_N * 4);
  unsigned int*       area    = (unsigned int*)alloc((size_t)HW_N * 4);
  unsigned long long* hash    = (unsigned long long*)alloc((size_t)HASH_SIZE * 8);
  unsigned char*      troot   = (unsigned char*)alloc((size_t)HW_N);
  unsigned char*      proot   = (unsigned char*)alloc((size_t)HW_N);
  unsigned char*      matched = (unsigned char*)alloc((size_t)HW_N);

  const int  TB    = 256;
  const int  gHW   = (HW_N + TB - 1) / TB;       // 4096 blocks
  const int  gHash = (HASH_SIZE + TB - 1) / TB;  // 8192 blocks

  k_clear_acc<<<1, 32, 0, stream>>>(acc);

  for (int img = 0; img < NIMG; ++img) {
    const float* lg = inputs + (size_t)img * HW_N;
    const float* tg = targets + (size_t)img * HW_N;

    k_clear_img<<<gHash, TB, 0, stream>>>(area, matched, hash);
    k_init_labels<<<gHW, TB, 0, stream>>>(lg, tg, Lp, Lt);
    k_merge<<<gHW, TB, 0, stream>>>(Lt);
    k_merge<<<gHW, TB, 0, stream>>>(Lp);
    k_compress<<<gHW, TB, 0, stream>>>(Lt, troot);
    k_compress<<<gHW, TB, 0, stream>>>(Lp, proot);
    k_stats<<<gHW, TB, 0, stream>>>(Lt, Lp, area, hash);
    k_match<<<gHash, TB, 0, stream>>>(hash, area, matched);

    // 128 blocks * 256 threads = 1024 waves -> exactly one 1024B chunk each.
    k_count_wmma<<<128, TB, 0, stream>>>(matched, &acc[img]);
    k_count_wmma<<<128, TB, 0, stream>>>(troot,   &acc[NIMG + img]);
    k_count_wmma<<<128, TB, 0, stream>>>(proot,   &acc[2 * NIMG + img]);
  }

  k_final<<<1, 1, 0, stream>>>(acc, out);
}
